// Label_correlation_Graph_82386062672094
// MI455X (gfx1250) — matrix operand order repaired and checked by reading, not compiled, not run on hardware
//
#include <hip/hip_runtime.h>
#include <hip/hip_bf16.h>

// ---------------------------------------------------------------------------
// Shapes: B=8, N=4096, D=512, H=128, M = B*N = 32768 rows per modality.
// All inputs f32. WMMA path uses f16 A/B with f32 accumulation.
// ---------------------------------------------------------------------------

typedef _Float16 v16h __attribute__((ext_vector_type(16)));
typedef float    v8f  __attribute__((ext_vector_type(8)));

// Async global->LDS copies (ASYNCcnt path), if the toolchain exposes them.
// Probe-confirmed parameter types (from hipcc diagnostics):
//   b128: (v4i as1*, v4i as3*, imm int, imm int)
//   b32 : (int as1*, int as3*, imm int, imm int)
#if __has_builtin(__builtin_amdgcn_global_load_async_to_lds_b128) && \
    __has_builtin(__builtin_amdgcn_global_load_async_to_lds_b32) &&  \
    __has_builtin(__builtin_amdgcn_s_wait_asynccnt)
#define USE_ASYNC_LDS 1
typedef int v4i_t __attribute__((vector_size(16)));
typedef __attribute__((address_space(1))) v4i_t* gv4i_p;
typedef __attribute__((address_space(3))) v4i_t* lv4i_p;
typedef __attribute__((address_space(1))) int*   gi32_p;
typedef __attribute__((address_space(3))) int*   li32_p;
#else
#define USE_ASYNC_LDS 0
#endif

// D = A*B + C, 16x16x32 f16 -> f32  (CDNA5 v_wmma_f32_16x16x32_f16)
__device__ __forceinline__ v8f wmma_f16(const v16h a, const v16h b, const v8f c) {
  return __builtin_amdgcn_wmma_f32_16x16x32_f16(
      /*neg_a=*/false, a, /*neg_b=*/false, b,
      /*c_mod=*/(short)0, c, /*reuse_a=*/false, /*reuse_b=*/false);
}

// A-fragment (16x32 f16) from a row-major [16][512] LDS tile, K-tile kt.
// ISA layout: lanes 0-15 hold row M=lane, elems 0..7 = K kt*32+0..7,
// elems 8..15 = K kt*32+16..23; lanes 16-31: row M=lane-16, K offsets +8 / +24.
__device__ __forceinline__ v16h load_a_frag(const _Float16* tile, int lane, int kt) {
  const int row = lane & 15;
  const int kb0 = kt * 32 + ((lane < 16) ? 0 : 8);
  const _Float16* p0 = tile + row * 512 + kb0;
  const _Float16* p1 = p0 + 16;
  v16h a;
#pragma unroll
  for (int e = 0; e < 8; ++e) { a[e] = p0[e]; a[8 + e] = p1[e]; }
  return a;
}

// ---------------------------------------------------------------------------
// K0: convert weights to f16 in WMMA B-fragment-major order.
// B-fragment (32x16): lane l<16 -> col n = l, elems 0..7 -> K = 0..7, 8..15 -> K = 16..23 (rel);
// lane l>=16 -> col n = l-16, K offsets +8 / +24.
// Fragment f stored as 512 contiguous halves: off = f*512 + l*16 + e.
// gc1_w: [512][512], 16 K-tiles x 32 N-tiles (f = kt*32+nt).
// attV/U_w: [3][512][128], 16 K-tiles x 8 N-tiles (f = kt*8+nt) per m.
// ---------------------------------------------------------------------------
__global__ __launch_bounds__(256) void k0_prep(
    const float* __restrict__ gc1_w,
    const float* __restrict__ attV_w, const float* __restrict__ attU_w,
    _Float16* __restrict__ Wfrag, _Float16* __restrict__ Vfrag,
    _Float16* __restrict__ Ufrag) {
  const int tid = blockIdx.x * 256 + threadIdx.x;
  if (tid < 262144) {                       // gc1_w
    const int f = tid >> 9, r = tid & 511, l = r >> 4, e = r & 15;
    const int kt = f >> 5, nt = f & 31;
    const int k = kt * 32 + ((l < 16) ? 0 : 8) + (e & 7) + ((e >> 3) << 4);
    const int n = nt * 16 + (l & 15);
    Wfrag[tid] = (_Float16)gc1_w[k * 512 + n];
  } else if (tid < 262144 + 2 * 196608) {   // attV_w / attU_w
    const int which = (tid - 262144) / 196608;  // 0=V, 1=U
    const int t2 = (tid - 262144) % 196608;
    const int m = t2 >> 16, rem = t2 & 65535;
    const int f = rem >> 9, r = rem & 511, l = r >> 4, e = r & 15;
    const int kt = f >> 3, nt = f & 7;
    const int k = kt * 32 + ((l < 16) ? 0 : 8) + (e & 7) + ((e >> 3) << 4);
    const int n = nt * 16 + (l & 15);
    if (which == 0) Vfrag[t2] = (_Float16)attV_w[m * 65536 + k * 128 + n];
    else            Ufrag[t2] = (_Float16)attU_w[m * 65536 + k * 128 + n];
  }
}

// ---------------------------------------------------------------------------
// K1: per 16-row tile: GEMM1 (3 modalities, 16x512x512, WMMA) + ADJ mix
//     + relu/alpha blend + LayerNorm -> hn16 (f16) to global workspace.
// Block = 256 threads = 8 waves; wave w owns N-tiles {4w..4w+3}.
// (x tiles need f32->f16 conversion in flight, so no async copy here.)
// ---------------------------------------------------------------------------
__global__ __launch_bounds__(256) void k1_gcn_ln(
    const float* __restrict__ x0, const float* __restrict__ x1,
    const float* __restrict__ x2,
    const _Float16* __restrict__ Wfrag, const float* __restrict__ gc1_b,
    const float* __restrict__ lnS, const float* __restrict__ lnB,
    _Float16* __restrict__ hn16) {
  __shared__ _Float16 xA[3][16][512];           // 48 KB
  __shared__ float sSum[3][16], sSq[3][16], sMu[3][16], sRs[3][16];

  const int t = threadIdx.x;
  const int lane = t & 31, wv = t >> 5;
  const int row0 = blockIdx.x * 16;

  if (t < 48) { (&sSum[0][0])[t] = 0.f; (&sSq[0][0])[t] = 0.f; }

  // Load x tiles, convert to f16.
  {
    const float* srcs[3] = {x0, x1, x2};
#pragma unroll
    for (int m = 0; m < 3; ++m) {
      const float* s = srcs[m] + (size_t)row0 * 512;
      _Float16* dst = &xA[m][0][0];
      for (int i = t; i < 8192; i += 256) dst[i] = (_Float16)s[i];
    }
  }
  __syncthreads();

  // ---- GEMM1: 3 modalities x 4 N-tiles of accumulators ----
  v8f acc[3][4];
#pragma unroll
  for (int m = 0; m < 3; ++m)
#pragma unroll
    for (int j = 0; j < 4; ++j)
#pragma unroll
      for (int r = 0; r < 8; ++r) acc[m][j][r] = 0.f;

  for (int kt = 0; kt < 16; ++kt) {
    v16h af[3];
#pragma unroll
    for (int m = 0; m < 3; ++m) af[m] = load_a_frag(&xA[m][0][0], lane, kt);
    if (kt < 15)
      __builtin_prefetch(Wfrag + (size_t)((kt + 1) * 32 + wv * 4) * 512 + lane * 16, 0, 1);
#pragma unroll
    for (int j = 0; j < 4; ++j) {
      const int nt = wv * 4 + j;
      const v16h bf = *(const v16h*)(Wfrag + (size_t)(kt * 32 + nt) * 512 + lane * 16);
      acc[0][j] = wmma_f16(af[0], bf, acc[0][j]);
      acc[1][j] = wmma_f16(af[1], bf, acc[1][j]);
      acc[2][j] = wmma_f16(af[2], bf, acc[2][j]);
    }
  }

  // ---- ADJ mix + bias + relu + alpha blend; accumulate LN stats ----
  const float A00 = 1.0f, A01 = 0.4038f, A02 = 0.3035f;
  const float A10 = 1.0f, A11 = 1.0f,    A12 = 0.1263f;
  const float A20 = 0.2595f, A21 = 0.0436f, A22 = 1.0f;
  const float R0 = A00 + A01 + A02, R1 = A10 + A11 + A12, R2 = A20 + A21 + A22;

  float psum[3][8], psq[3][8];
#pragma unroll
  for (int i = 0; i < 3; ++i)
#pragma unroll
    for (int r = 0; r < 8; ++r) { psum[i][r] = 0.f; psq[i][r] = 0.f; }

#pragma unroll
  for (int j = 0; j < 4; ++j) {
    const int col = (wv * 4 + j) * 16 + (lane & 15);
    const float gb = gc1_b[col];
#pragma unroll
    for (int r = 0; r < 8; ++r) {
      const int row = (lane < 16) ? r : r + 8;
      const float s0 = acc[0][j][r], s1 = acc[1][j][r], s2 = acc[2][j][r];
      const float g0 = A00 * s0 + A01 * s1 + A02 * s2 + R0 * gb;
      const float g1 = A10 * s0 + A11 * s1 + A12 * s2 + R1 * gb;
      const float g2 = A20 * s0 + A21 * s1 + A22 * s2 + R2 * gb;
      const float m0 = fmaxf(g0, 0.f) * 0.5f + (float)xA[0][row][col] * 0.5f;
      const float m1 = fmaxf(g1, 0.f) * 0.5f + (float)xA[1][row][col] * 0.5f;
      const float m2 = fmaxf(g2, 0.f) * 0.5f + (float)xA[2][row][col] * 0.5f;
      acc[0][j][r] = m0; acc[1][j][r] = m1; acc[2][j][r] = m2;
      psum[0][r] += m0; psq[0][r] += m0 * m0;
      psum[1][r] += m1; psq[1][r] += m1 * m1;
      psum[2][r] += m2; psq[2][r] += m2 * m2;
    }
  }

  // Reduce partials across each 16-lane half, then into LDS.
#pragma unroll
  for (int i = 0; i < 3; ++i)
#pragma unroll
    for (int r = 0; r < 8; ++r) {
      float s = psum[i][r], q = psq[i][r];
      s += __shfl_xor(s, 1, 16); q += __shfl_xor(q, 1, 16);
      s += __shfl_xor(s, 2, 16); q += __shfl_xor(q, 2, 16);
      s += __shfl_xor(s, 4, 16); q += __shfl_xor(q, 4, 16);
      s += __shfl_xor(s, 8, 16); q += __shfl_xor(q, 8, 16);
      if ((lane & 15) == 0) {
        const int row = (lane < 16) ? r : r + 8;
        atomicAdd(&sSum[i][row], s);
        atomicAdd(&sSq[i][row], q);
      }
    }
  __syncthreads();
  if (t < 48) {
    const int i = t / 16, row = t % 16;
    const float mu = sSum[i][row] * (1.f / 512.f);
    const float var = sSq[i][row] * (1.f / 512.f) - mu * mu;
    sMu[i][row] = mu;
    sRs[i][row] = rsqrtf(var + 1e-6f);
  }
  __syncthreads();

  // ---- Normalize and store hn as f16 ----
#pragma unroll
  for (int j = 0; j < 4; ++j) {
    const int col = (wv * 4 + j) * 16 + (lane & 15);
#pragma unroll
    for (int r = 0; r < 8; ++r) {
      const int row = (lane < 16) ? r : r + 8;
#pragma unroll
      for (int i = 0; i < 3; ++i) {
        const float hn = (acc[i][j][r] - sMu[i][row]) * sRs[i][row] *
                             lnS[i * 512 + col] + lnB[i * 512 + col];
        hn16[((size_t)i * 32768 + row0 + row) * 512 + col] = (_Float16)hn;
      }
    }
  }
}

// ---------------------------------------------------------------------------
// K2: per (tile, m): gated attention pre-softmax scalar s[row] and
//     t[c][row] = hn . fc_w[m][c]  (for saliency outputs).
// hn tile is a verbatim f16 copy -> async global->LDS path when available.
// Wave w owns H N-tile w (8 tiles of 16 = H=128). Two WMMA accumulators (V,U).
// ---------------------------------------------------------------------------
__global__ __launch_bounds__(256) void k2_att(
    const _Float16* __restrict__ hn16,
    const _Float16* __restrict__ Vfrag, const _Float16* __restrict__ Ufrag,
    const float* __restrict__ attVb, const float* __restrict__ attUb,
    const float* __restrict__ attw, const float* __restrict__ attwb,
    const float* __restrict__ fcw,
    float* __restrict__ sArr, float* __restrict__ tArr) {
  __shared__ _Float16 hA[16][512];              // 16 KB
  __shared__ float sS[16];
  const int t = threadIdx.x, lane = t & 31, wv = t >> 5;
  const int tile = blockIdx.x, m = blockIdx.y;
  const int gRow0 = tile * 16;

  if (t < 16) sS[t] = 0.f;
  {
    const _Float16* src = hn16 + ((size_t)m * 32768 + gRow0) * 512;
    _Float16* dst = &hA[0][0];
#if USE_ASYNC_LDS
    // 16 KB = 1024 x 16B chunks; each thread issues 4 async B128 copies.
#pragma unroll
    for (int i = t; i < 1024; i += 256) {
      __builtin_amdgcn_global_load_async_to_lds_b128(
          (gv4i_p)(src + i * 8), (lv4i_p)(dst + i * 8), 0, 0);
    }
    __builtin_amdgcn_s_wait_asynccnt(0);
#else
    for (int i = t; i < 8192; i += 256) dst[i] = src[i];
#endif
  }
  __syncthreads();

  v8f accV, accU;
#pragma unroll
  for (int r = 0; r < 8; ++r) { accV[r] = 0.f; accU[r] = 0.f; }

  const size_t mo = (size_t)m * 65536;
  for (int kt = 0; kt < 16; ++kt) {
    const v16h a = load_a_frag(&hA[0][0], lane, kt);
    const size_t fo = mo + (size_t)(kt * 8 + wv) * 512 + lane * 16;
    const v16h bv = *(const v16h*)(Vfrag + fo);
    const v16h bu = *(const v16h*)(Ufrag + fo);
    accV = wmma_f16(a, bv, accV);
    accU = wmma_f16(a, bu, accU);
  }

  // gate = tanh(V+b) * sigmoid(U+b); s[row] += gate . attw
  const int hcol = wv * 16 + (lane & 15);
  const float vb = attVb[m * 128 + hcol];
  const float ub = attUb[m * 128 + hcol];
  const float aw = attw[m * 128 + hcol];
#pragma unroll
  for (int r = 0; r < 8; ++r) {
    const int row = (lane < 16) ? r : r + 8;
    const float g = tanhf(accV[r] + vb) * (1.f / (1.f + __expf(-(accU[r] + ub))));
    float val = g * aw;
    val += __shfl_xor(val, 1, 16);
    val += __shfl_xor(val, 2, 16);
    val += __shfl_xor(val, 4, 16);
    val += __shfl_xor(val, 8, 16);
    if ((lane & 15) == 0) atomicAdd(&sS[row], val);
  }

  // t[c][row] = hn . fc_w[m][c] — wave w handles rows 2w, 2w+1.
#pragma unroll
  for (int rr = 0; rr < 2; ++rr) {
    const int row = wv * 2 + rr;
    float t0 = 0.f, t1 = 0.f;
    for (int d = lane; d < 512; d += 32) {
      const float hv = (float)hA[row][d];
      t0 += hv * fcw[m * 1024 + d];
      t1 += hv * fcw[m * 1024 + 512 + d];
    }
#pragma unroll
    for (int msk = 16; msk >= 1; msk >>= 1) {
      t0 += __shfl_xor(t0, msk, 32);
      t1 += __shfl_xor(t1, msk, 32);
    }
    if (lane == 0) {
      const int gRow = gRow0 + row;
      tArr[(m * 2 + 0) * 32768 + gRow] = t0;
      tArr[(m * 2 + 1) * 32768 + gRow] = t1;
    }
  }
  __syncthreads();
  if (t < 16) sArr[m * 32768 + gRow0 + t] = sS[t] + attwb[m];
}

// ---------------------------------------------------------------------------
// K3: softmax over N per (m,b); write a[] and the 4 saliency outputs.
// ---------------------------------------------------------------------------
__global__ __launch_bounds__(256) void k3_softmax_sal(
    const float* __restrict__ sArr, const float* __restrict__ tArr,
    const float* __restrict__ fcb, float* __restrict__ aArr,
    float* __restrict__ out) {
  __shared__ float red[256];
  const int t = threadIdx.x;
  const int m = blockIdx.x >> 3, b = blockIdx.x & 7;
  const int base = m * 32768 + b * 4096;

  float mx = -3.4e38f;
  for (int n = t; n < 4096; n += 256) mx = fmaxf(mx, sArr[base + n]);
  red[t] = mx; __syncthreads();
  for (int s = 128; s > 0; s >>= 1) {
    if (t < s) red[t] = fmaxf(red[t], red[t + s]);
    __syncthreads();
  }
  const float gmax = red[0]; __syncthreads();

  float ls = 0.f;
  for (int n = t; n < 4096; n += 256) ls += __expf(sArr[base + n] - gmax);
  red[t] = ls; __syncthreads();
  for (int s = 128; s > 0; s >>= 1) {
    if (t < s) red[t] += red[t + s];
    __syncthreads();
  }
  const float inv = 1.f / red[0];

  for (int n = t; n < 4096; n += 256) {
    const float a = __expf(sArr[base + n] - gmax) * inv;
    aArr[base + n] = a;
    const int bn = b * 4096 + n;
    if (m == 0) {
      out[48 + bn]    = a * tArr[bn]             + fcb[0] * (1.f / 4096.f);
      out[32816 + bn] = a * tArr[32768 + bn]     + fcb[1] * (1.f / 4096.f);
    } else if (m == 1) {
      out[65584 + bn] = a * tArr[3 * 32768 + bn] + fcb[3] * (1.f / 4096.f);
    } else {
      out[98352 + bn] = a * tArr[5 * 32768 + bn] + fcb[5] * (1.f / 4096.f);
    }
  }
}

// ---------------------------------------------------------------------------
// K4: enc[m][b][d] = sum_n a * hn; writes encs + mark_output.
// The a-chunk staging is a verbatim copy -> async global->LDS when available.
// ---------------------------------------------------------------------------
__global__ __launch_bounds__(256) void k4_enc(
    const _Float16* __restrict__ hn16, const float* __restrict__ aArr,
    float* __restrict__ encArr, float* __restrict__ out) {
  __shared__ float sA[256];
  const int t = threadIdx.x;
  const int m = blockIdx.x >> 3, b = blockIdx.x & 7;
  const _Float16* hb = hn16 + ((size_t)m * 32768 + b * 4096) * 512;
  const float* ab = aArr + m * 32768 + b * 4096;
  float a0 = 0.f, a1 = 0.f;
  for (int n0 = 0; n0 < 4096; n0 += 256) {
#if USE_ASYNC_LDS
    __builtin_amdgcn_global_load_async_to_lds_b32(
        (gi32_p)(ab + n0 + t), (li32_p)(&sA[t]), 0, 0);
    __builtin_amdgcn_s_wait_asynccnt(0);
#else
    sA[t] = ab[n0 + t];
#endif
    __syncthreads();
    if (n0 + 256 < 4096)
      __builtin_prefetch(hb + (size_t)(n0 + 256) * 512 + t, 0, 1);
    for (int k = 0; k < 256; ++k) {
      const _Float16* hr = hb + (size_t)(n0 + k) * 512;
      const float av = sA[k];
      a0 += av * (float)hr[t];
      a1 += av * (float)hr[t + 256];
    }
    __syncthreads();
  }
  encArr[(m * 8 + b) * 512 + t] = a0;
  encArr[(m * 8 + b) * 512 + t + 256] = a1;
  out[131120 + m * 4096 + b * 512 + t] = a0;
  out[131120 + m * 4096 + b * 512 + t + 256] = a1;
  const int slot = (m == 0) ? 0 : ((m == 1) ? 2 : 1);  // mark = [enc0, enc2, enc1]
  out[143408 + b * 1536 + slot * 512 + t] = a0;
  out[143408 + b * 1536 + slot * 512 + t + 256] = a1;
}

// ---------------------------------------------------------------------------
// K5: logits[m][b][c] = enc . fc_w[m][c] + fc_b[m][c]
// ---------------------------------------------------------------------------
__global__ __launch_bounds__(32) void k5_logits(
    const float* __restrict__ encArr, const float* __restrict__ fcw,
    const float* __restrict__ fcb, float* __restrict__ out) {
  const int lane = threadIdx.x;
  const int id = blockIdx.x;
  const int m = id >> 4, rr = id & 15, b = rr >> 1, c = rr & 1;
  const float* e = encArr + (m * 8 + b) * 512;
  const float* w = fcw + m * 1024 + c * 512;
  float p = 0.f;
  for (int d = lane; d < 512; d += 32) p += e[d] * w[d];
#pragma unroll
  for (int msk = 16; msk >= 1; msk >>= 1) p += __shfl_xor(p, msk, 32);
  if (lane == 0) out[m * 16 + b * 2 + c] = p + fcb[m * 2 + c];
}

// ---------------------------------------------------------------------------
extern "C" void kernel_launch(void* const* d_in, const int* in_sizes, int n_in,
                              void* d_out, int out_size, void* d_ws,
                              size_t ws_size, hipStream_t stream) {
  const float* x0    = (const float*)d_in[0];
  const float* x1    = (const float*)d_in[1];
  const float* x2    = (const float*)d_in[2];
  const float* gc1_w = (const float*)d_in[3];
  const float* gc1_b = (const float*)d_in[4];
  const float* lnS   = (const float*)d_in[5];
  const float* lnB   = (const float*)d_in[6];
  const float* attVw = (const float*)d_in[7];
  const float* attVb = (const float*)d_in[8];
  const float* attUw = (const float*)d_in[9];
  const float* attUb = (const float*)d_in[10];
  const float* attw  = (const float*)d_in[11];
  const float* attwb = (const float*)d_in[12];
  const float* fcw   = (const float*)d_in[13];
  const float* fcb   = (const float*)d_in[14];
  float* out = (float*)d_out;

  char* ws = (char*)d_ws;
  _Float16* Wfrag  = (_Float16*)(ws + 0);          //   512 KB
  _Float16* Vfrag  = (_Float16*)(ws + 524288);     //   384 KB
  _Float16* Ufrag  = (_Float16*)(ws + 917504);     //   384 KB
  _Float16* hn16   = (_Float16*)(ws + 1310720);    //    96 MB
  float*    sArr   = (float*)(ws + 101974016);     //   384 KB
  float*    tArr   = (float*)(ws + 102367232);     //   768 KB
  float*    aArr   = (float*)(ws + 103153664);     //   384 KB
  float*    encArr = (float*)(ws + 103546880);     //    48 KB

  k0_prep<<<2560, 256, 0, stream>>>(gc1_w, attVw, attUw, Wfrag, Vfrag, Ufrag);
  k1_gcn_ln<<<2048, 256, 0, stream>>>(x0, x1, x2, Wfrag, gc1_b, lnS, lnB, hn16);
  k2_att<<<dim3(2048, 3), 256, 0, stream>>>(hn16, Vfrag, Ufrag, attVb, attUb,
                                            attw, attwb, fcw, sArr, tArr);
  k3_softmax_sal<<<24, 256, 0, stream>>>(sArr, tArr, fcb, aArr, out);
  k4_enc<<<24, 256, 0, stream>>>(hn16, aArr, encArr, out);
  k5_logits<<<48, 32, 0, stream>>>(encArr, fcw, fcb, out);
}
